// MultiHeadAttention_55542517071982
// MI455X (gfx1250) — matrix operand order repaired and checked by reading, not compiled, not run on hardware
//
#include <hip/hip_runtime.h>

typedef __bf16 bf16_t;
typedef __attribute__((ext_vector_type(16))) __bf16 v16bf;
typedef __attribute__((ext_vector_type(8)))  float  v8f;
typedef __attribute__((ext_vector_type(4)))  int    v4i;

#define DM    1024
#define HEADS 16
#define DH    64
#define SEQ   2048
#define BATCH 4
#define MTOT  (BATCH * SEQ)   // 8192

// ---------- helpers ----------------------------------------------------------

static __device__ __forceinline__ unsigned pack2bf(float a, float b) {
  unsigned ua = __builtin_bit_cast(unsigned, a);
  unsigned ub = __builtin_bit_cast(unsigned, b);
  ua += 0x7FFFu + ((ua >> 16) & 1u);   // round-to-nearest-even
  ub += 0x7FFFu + ((ub >> 16) & 1u);
  return (ua >> 16) | (ub & 0xFFFF0000u);
}

static __device__ __forceinline__ bf16_t f2bf(float a) {
  unsigned ua = __builtin_bit_cast(unsigned, a);
  ua += 0x7FFFu + ((ua >> 16) & 1u);
  unsigned short h = (unsigned short)(ua >> 16);
  return __builtin_bit_cast(bf16_t, h);
}

union FragU { uint4 u[2]; v16bf v; };

static __device__ __forceinline__ v8f zero8() {
  v8f z;
  for (int i = 0; i < 8; i++) z[i] = 0.0f;
  return z;
}

// ---- async global -> LDS staging (CDNA5 ASYNCcnt path), with safe fallback --
#if __has_builtin(__builtin_amdgcn_global_load_async_to_lds_b128)
#define HAS_ASYNC_LDS 1
#else
#define HAS_ASYNC_LDS 0
#endif

// copy 16 bytes (8 bf16) global -> LDS
static __device__ __forceinline__ void stage_b128(const bf16_t* g, bf16_t* l) {
#if HAS_ASYNC_LDS
  __builtin_amdgcn_global_load_async_to_lds_b128(
      (__attribute__((address_space(1))) v4i*)(bf16_t*)g,
      (__attribute__((address_space(3))) v4i*)l, 0, 0);
#else
  *(uint4*)l = *(const uint4*)g;
#endif
}

static __device__ __forceinline__ void stage_wait() {
#if HAS_ASYNC_LDS
#if __has_builtin(__builtin_amdgcn_s_wait_asynccnt)
  __builtin_amdgcn_s_wait_asynccnt(0);
#else
  asm volatile("s_wait_asynccnt 0x0" ::: "memory");
#endif
#endif
}

// A-fragment (16x32 bf16, M x K): lane L holds row (L&15).
// lanes 0-15: VGPR0-3 = K k0..k0+7, VGPR4-7 = K k0+16..k0+23
// lanes 16-31: VGPR0-3 = K k0+8..k0+15, VGPR4-7 = K k0+24..k0+31
static __device__ __forceinline__ v16bf frag_a(const bf16_t* row0,
                                               int stride_bytes, int lane, int k0) {
  int lo = lane & 15, hi = lane >> 4;
  const char* p = (const char*)row0 + lo * stride_bytes;
  FragU f;
  f.u[0] = *(const uint4*)(p + (k0 + hi * 8) * 2);
  f.u[1] = *(const uint4*)(p + (k0 + 16 + hi * 8) * 2);
  return f.v;
}

// B-fragment (32x16 bf16, K x N): lane L holds column (L&15).
// lanes 0-15: K k0..k0+7, k0+16..k0+23; lanes 16-31: K k0+8..k0+15, k0+24..k0+31.
// LDS tile must be stored [n][k] (K contiguous per column-row).
static __device__ __forceinline__ v16bf frag_b(const bf16_t* row0,
                                               int stride_bytes, int lane, int k0) {
  int lo = lane & 15, hi = lane >> 4;
  const char* p = (const char*)row0 + lo * stride_bytes;
  FragU f;
  f.u[0] = *(const uint4*)(p + (k0 + hi * 8) * 2);
  f.u[1] = *(const uint4*)(p + (k0 + 16 + hi * 8) * 2);
  return f.v;
}

static __device__ __forceinline__ v8f wmma_bf16(v16bf a, v16bf b, v8f c) {
  return __builtin_amdgcn_wmma_f32_16x16x32_bf16(false, a, false, b,
                                                 (short)0, c, false, false);
}

// ---------- GEMM: [8192 x 1024] @ [1024 x 1024] + bias -----------------------
// A_BF16: A operand is bf16 (ctx) vs f32 (x).  OUT_MODE 0: bf16 out in
// [B,H,S,Dh] layout (QKV projections); OUT_MODE 1: f32 flat [M, DM].
template <bool A_BF16, int OUT_MODE>
__global__ __launch_bounds__(256) void gemm128(const void* __restrict__ Av,
                                               const float* __restrict__ W,
                                               const float* __restrict__ bias,
                                               void* __restrict__ Ov) {
  __shared__ __align__(16) bf16_t As[128 * 40];  // [m][k], stride 40
  __shared__ __align__(16) bf16_t Bs[128 * 40];  // [n][k], stride 40 (W transposed)

  const int t    = threadIdx.x;
  const int lane = t & 31;
  const int wid  = t >> 5;
  const int wm   = wid >> 1;           // 0..3 -> 32-row strip
  const int wn   = wid & 1;            // 0..1 -> 64-col strip
  const int M0   = blockIdx.y * 128;
  const int N0   = blockIdx.x * 128;

  v8f acc[2][4];
  for (int i = 0; i < 2; i++)
    for (int j = 0; j < 4; j++) acc[i][j] = zero8();

  for (int kb = 0; kb < DM; kb += 32) {
    __syncthreads();
    // ---- A tile: 128 x 32 -> bf16 LDS (row-major)
    {
      int row = t >> 1, half = t & 1;
      if (A_BF16) {
        // already bf16: use the async global->LDS path (no VGPR round-trip)
        const bf16_t* A = (const bf16_t*)Av;
        const bf16_t* g = A + (size_t)(M0 + row) * DM + kb + half * 16;
        bf16_t* l = &As[row * 40 + half * 16];
        stage_b128(g, l);
        stage_b128(g + 8, l + 8);
      } else {
        const float* A = (const float*)Av;
        const float4* g = (const float4*)(A + (size_t)(M0 + row) * DM + kb + half * 16);
        float4 f0 = g[0], f1 = g[1], f2 = g[2], f3 = g[3];
        uint4 p0, p1;
        p0.x = pack2bf(f0.x, f0.y); p0.y = pack2bf(f0.z, f0.w);
        p0.z = pack2bf(f1.x, f1.y); p0.w = pack2bf(f1.z, f1.w);
        p1.x = pack2bf(f2.x, f2.y); p1.y = pack2bf(f2.z, f2.w);
        p1.z = pack2bf(f3.x, f3.y); p1.w = pack2bf(f3.z, f3.w);
        uint4* s = (uint4*)&As[row * 40 + half * 16];
        s[0] = p0; s[1] = p1;
      }
    }
    // ---- W tile: 32 x 128 -> K-transposed bf16 LDS Bs[n][k]
    for (int u = t; u < 512; u += 256) {
      int kp = u & 15;         // k-pair 0..15
      int nc = u >> 4;         // n-chunk of 4, 0..31
      int k = kp * 2, n = nc * 4;
      const float* w0 = W + (size_t)(kb + k) * DM + N0 + n;
      float4 r0 = *(const float4*)w0;
      float4 r1 = *(const float4*)(w0 + DM);
      *(unsigned*)&Bs[(n + 0) * 40 + k] = pack2bf(r0.x, r1.x);
      *(unsigned*)&Bs[(n + 1) * 40 + k] = pack2bf(r0.y, r1.y);
      *(unsigned*)&Bs[(n + 2) * 40 + k] = pack2bf(r0.z, r1.z);
      *(unsigned*)&Bs[(n + 3) * 40 + k] = pack2bf(r0.w, r1.w);
    }
    // hint: prefetch next K-step tiles (global_prefetch_b8)
    if (kb + 32 < DM) {
      int row = t >> 1;
      __builtin_prefetch((const char*)Av +
                         ((size_t)(M0 + row) * DM + kb + 32) * (A_BF16 ? 2 : 4), 0, 3);
      __builtin_prefetch(W + (size_t)(kb + 32 + lane) * DM + N0 + wid * 16, 0, 3);
    }
    if (A_BF16) stage_wait();
    __syncthreads();
    // ---- 8 WMMAs per wave for this K-step
    for (int mi = 0; mi < 2; mi++) {
      v16bf a = frag_a(&As[(wm * 32 + mi * 16) * 40], 80, lane, 0);
      for (int ni = 0; ni < 4; ni++) {
        v16bf b = frag_b(&Bs[(wn * 64 + ni * 16) * 40], 80, lane, 0);
        acc[mi][ni] = wmma_bf16(a, b, acc[mi][ni]);
      }
    }
  }

  // ---- epilogue: bias + store (C layout: VGPR r -> row r / r+8, col = lane&15)
  const int lo = lane & 15, hi = lane >> 4;
  for (int mi = 0; mi < 2; mi++) {
    for (int ni = 0; ni < 4; ni++) {
      int n = N0 + wn * 64 + ni * 16 + lo;
      float bb = bias[n];
      int mbase = M0 + wm * 32 + mi * 16 + hi * 8;
      if (OUT_MODE == 0) {
        bf16_t* O = (bf16_t*)Ov;
        int h = n >> 6, d = n & 63;
        for (int r = 0; r < 8; r++) {
          int m = mbase + r;
          int b = m >> 11, s = m & 2047;
          O[(((size_t)(b * HEADS + h)) * SEQ + s) * DH + d] = f2bf(acc[mi][ni][r] + bb);
        }
      } else {
        float* O = (float*)Ov;
        for (int r = 0; r < 8; r++) {
          int m = mbase + r;
          O[(size_t)m * DM + n] = acc[mi][ni][r] + bb;
        }
      }
    }
  }
}

// ---------- flash attention per (b,h), 128 q-rows per block ------------------
__global__ __launch_bounds__(256) void attn_kernel(const bf16_t* __restrict__ Q,
                                                   const bf16_t* __restrict__ K,
                                                   const bf16_t* __restrict__ V,
                                                   bf16_t* __restrict__ Ctx) {
  __shared__ __align__(16) bf16_t Qs[128 * 72];      // [q][d]
  __shared__ __align__(16) bf16_t Ks[64 * 72];       // [key][d]  (B-frag ready)
  __shared__ __align__(16) bf16_t Vt[64 * 72];       // [d][key]  (B-frag ready)
  __shared__ __align__(16) bf16_t Ps[8 * 16 * 72];   // per-wave P [q][key]

  const int t    = threadIdx.x;
  const int lane = t & 31;
  const int wid  = t >> 5;
  const int lo   = lane & 15, hi = lane >> 4;
  const int bh   = blockIdx.x;            // b*16 + h
  const int qb   = blockIdx.y;            // q block
  const int b    = bh >> 4, h = bh & 15;

  const size_t head_off = (size_t)bh * SEQ * DH;
  const bf16_t* Qg = Q + head_off + (size_t)qb * 128 * DH;
  const bf16_t* Kg = K + head_off;
  const bf16_t* Vg = V + head_off;

  // stage Q block 128x64 bf16 via async global->LDS
  {
    int row = t >> 1, half = t & 1;
    const bf16_t* g = Qg + row * DH + half * 32;
    bf16_t* l = &Qs[row * 72 + half * 32];
    stage_b128(g, l);
    stage_b128(g + 8, l + 8);
  }
  stage_wait();
  __syncthreads();
  // Q fragments live in registers for the whole pass (K-dim = 64 -> 2 frags)
  v16bf qf0 = frag_a(&Qs[(wid * 16) * 72], 144, lane, 0);
  v16bf qf1 = frag_a(&Qs[(wid * 16) * 72], 144, lane, 32);

  v8f o[4];
  float rmax[8], rsum[8];
  for (int j = 0; j < 4; j++) o[j] = zero8();
  for (int r = 0; r < 8; r++) { rmax[r] = -1e30f; rsum[r] = 0.0f; }

  bf16_t* Pw = &Ps[wid * 16 * 72];

  for (int kb = 0; kb < SEQ; kb += 64) {
    __syncthreads();
    // K tile 64x64 (straight copy, async global->LDS)
    {
      int row = t >> 2, q4 = t & 3;
      stage_b128(Kg + (size_t)(kb + row) * DH + q4 * 16, &Ks[row * 72 + q4 * 16]);
    }
    // V tile 64x64, transposed into Vt[d][key] (key-pairs packed as b32)
    {
      int kp = t >> 3, dc = t & 7;
      int key = kp * 2, d = dc * 8;
      uint4 v0 = *(const uint4*)(Vg + (size_t)(kb + key) * DH + d);
      uint4 v1 = *(const uint4*)(Vg + (size_t)(kb + key + 1) * DH + d);
      const unsigned short* a = (const unsigned short*)&v0;
      const unsigned short* c = (const unsigned short*)&v1;
      for (int i = 0; i < 8; i++) {
        unsigned pv = (unsigned)a[i] | ((unsigned)c[i] << 16);
        *(unsigned*)&Vt[(d + i) * 72 + key] = pv;
      }
    }
    if (kb + 64 < SEQ) {
      __builtin_prefetch(Kg + (size_t)(kb + 64 + (t >> 2)) * DH, 0, 3);
      __builtin_prefetch(Vg + (size_t)(kb + 64 + (t >> 2)) * DH, 0, 3);
    }
    stage_wait();
    __syncthreads();

    // scores S = (Q K^T) * 1/sqrt(Dh)   -> 16 x 64 per wave
    v8f sc[4];
    for (int j = 0; j < 4; j++) {
      v16bf b0 = frag_b(&Ks[(j * 16) * 72], 144, lane, 0);
      v16bf b1 = frag_b(&Ks[(j * 16) * 72], 144, lane, 32);
      v8f cacc = zero8();
      cacc = wmma_bf16(qf0, b0, cacc);
      cacc = wmma_bf16(qf1, b1, cacc);
      sc[j] = cacc * 0.125f;
    }

    // online softmax (row r lives in VGPR r across a 16-lane half)
    float mnew[8], cf[8];
    for (int r = 0; r < 8; r++) {
      float mv = fmaxf(fmaxf(sc[0][r], sc[1][r]), fmaxf(sc[2][r], sc[3][r]));
      for (int msk = 8; msk >= 1; msk >>= 1) mv = fmaxf(mv, __shfl_xor(mv, msk, 32));
      mnew[r] = fmaxf(rmax[r], mv);
      cf[r]   = __expf(rmax[r] - mnew[r]);
      rmax[r] = mnew[r];
    }
    for (int j = 0; j < 4; j++)
      for (int r = 0; r < 8; r++) sc[j][r] = __expf(sc[j][r] - mnew[r]);
    for (int r = 0; r < 8; r++) {
      float sv = sc[0][r] + sc[1][r] + sc[2][r] + sc[3][r];
      for (int msk = 8; msk >= 1; msk >>= 1) sv += __shfl_xor(sv, msk, 32);
      rsum[r] = rsum[r] * cf[r] + sv;
    }
    for (int j = 0; j < 4; j++)
      for (int r = 0; r < 8; r++) o[j][r] *= cf[r];

    // P -> per-wave LDS (C layout -> A layout re-stage)
    for (int j = 0; j < 4; j++)
      for (int r = 0; r < 8; r++)
        Pw[(r + hi * 8) * 72 + j * 16 + lo] = f2bf(sc[j][r]);

    v16bf p0 = frag_a(Pw, 144, lane, 0);
    v16bf p1 = frag_a(Pw, 144, lane, 32);
    for (int j = 0; j < 4; j++) {
      v16bf vb0 = frag_b(&Vt[(j * 16) * 72], 144, lane, 0);
      v16bf vb1 = frag_b(&Vt[(j * 16) * 72], 144, lane, 32);
      o[j] = wmma_bf16(p0, vb0, o[j]);
      o[j] = wmma_bf16(p1, vb1, o[j]);
    }
  }

  // finalize: ctx stored bf16 in [B, S, D_MODEL] layout for the output GEMM
  for (int j = 0; j < 4; j++) {
    for (int r = 0; r < 8; r++) {
      float val = o[j][r] / rsum[r];
      int s = qb * 128 + wid * 16 + r + hi * 8;
      int d = j * 16 + lo;
      Ctx[((size_t)(b * SEQ + s)) * DM + h * DH + d] = f2bf(val);
    }
  }
}

// ---------- launcher ---------------------------------------------------------

extern "C" void kernel_launch(void* const* d_in, const int* in_sizes, int n_in,
                              void* d_out, int out_size, void* d_ws, size_t ws_size,
                              hipStream_t stream) {
  (void)in_sizes; (void)n_in; (void)out_size; (void)ws_size;
  const float* x  = (const float*)d_in[0];
  const float* Wq = (const float*)d_in[1];
  const float* bq = (const float*)d_in[2];
  const float* Wk = (const float*)d_in[3];
  const float* bk = (const float*)d_in[4];
  const float* Wv = (const float*)d_in[5];
  const float* bv = (const float*)d_in[6];
  const float* Wo = (const float*)d_in[7];
  const float* bo = (const float*)d_in[8];

  const size_t qkv_elems = (size_t)BATCH * HEADS * SEQ * DH;  // 8,388,608
  bf16_t* qb_p  = (bf16_t*)d_ws;
  bf16_t* kb_p  = qb_p + qkv_elems;
  bf16_t* vb_p  = qb_p + 2 * qkv_elems;
  bf16_t* ctx_p = qb_p + 3 * qkv_elems;

  dim3 blk(256);
  dim3 gp(DM / 128, MTOT / 128);        // (8, 64)
  gemm128<false, 0><<<gp, blk, 0, stream>>>((const void*)x, Wq, bq, (void*)qb_p);
  gemm128<false, 0><<<gp, blk, 0, stream>>>((const void*)x, Wk, bk, (void*)kb_p);
  gemm128<false, 0><<<gp, blk, 0, stream>>>((const void*)x, Wv, bv, (void*)vb_p);

  dim3 ga(BATCH * HEADS, SEQ / 128);    // (64, 16)
  attn_kernel<<<ga, blk, 0, stream>>>(qb_p, kb_p, vb_p, ctx_p);

  dim3 go(DM / 128, MTOT / 128);
  gemm128<true, 1><<<go, blk, 0, stream>>>((const void*)ctx_p, Wo, bo, d_out);
}
